// crossAttention_19430432047352
// MI455X (gfx1250) — matrix operand order repaired and checked by reading, not compiled, not run on hardware
//
#include <hip/hip_runtime.h>
#include <hip/hip_bf16.h>
#include <math.h>
#include <stdint.h>

// ---------------------------------------------------------------------------
// Cross-attention for B=2, C=1024, D=64, N=128 on gfx1250 (CDNA5, wave32).
// All GEMMs run on v_wmma_f32_16x16x32_bf16 with f32 accumulation.
// V-tile staging uses GLOBAL_LOAD_ASYNC_TO_LDS_B128 (ASYNCcnt) via inline asm.
// ---------------------------------------------------------------------------

typedef __bf16 bf16_t;
typedef __attribute__((ext_vector_type(16))) bf16_t v16bf;
typedef __attribute__((ext_vector_type(8)))  float  v8f;

union FragU { v16bf v; unsigned u[8]; };

#define MDIM 1024   // output channels
#define KDIM 1024   // input channels
#define NPOS 8192   // D*N flattened positions per batch
#define DD   64
#define NN   128
#define ATT_SCALE 0.125f   // 64^-0.5

// ---------------------------------------------------------------------------
// Channel-projection GEMM: Y_b[o][p] = act( sum_c W[o][c] * X_b[c][p] )
// Block: 128 threads (4 waves), 128x128 output tile, wave = 64x64 subtile.
// LDS: A tile 128x32 bf16 (row-major, K contiguous), B tile stored
// transposed as Bs[n][k] so the B fragment also reads contiguous K.
// Row stride 40 bf16 = 80 B keeps every row 16B-aligned for b128 LDS ops.
// ---------------------------------------------------------------------------
template<typename TB, typename TOUT, bool GELU>
__global__ __launch_bounds__(128)
void proj_gemm(const float* __restrict__ W, const TB* __restrict__ X,
               TOUT* __restrict__ Y)
{
  __shared__ __align__(16) bf16_t As[128][40];   // [m][k]
  __shared__ __align__(16) bf16_t Bs[128][40];   // [n][k]

  const int b  = blockIdx.z;
  const int m0 = blockIdx.y * 128;
  const int n0 = blockIdx.x * 128;
  const TB*  Xb = X + (size_t)b * KDIM * NPOS;
  TOUT*      Yb = Y + (size_t)b * MDIM * NPOS;

  const int tid  = threadIdx.x;
  const int lane = tid & 31;
  const int wave = tid >> 5;
  const int wm   = (wave >> 1) * 64;
  const int wn   = (wave & 1) * 64;
  const int hi   = lane >> 4;     // 0: lanes 0-15, 1: lanes 16-31
  const int l15  = lane & 15;

  v8f acc[4][4];
  const v8f vzero = {0.f,0.f,0.f,0.f,0.f,0.f,0.f,0.f};
#pragma unroll
  for (int i = 0; i < 4; ++i)
#pragma unroll
    for (int j = 0; j < 4; ++j) acc[i][j] = vzero;

  for (int kk = 0; kk < KDIM; kk += 32) {
    // ---- stage A tile: W[m0+t][kk..kk+32) -> bf16, 4x ds_store_b128
    {
      const float* src = W + (size_t)(m0 + tid) * KDIM + kk;
      if (kk + 32 < KDIM) __builtin_prefetch(src + 32, 0, 0);
      __align__(16) bf16_t ta[32];
#pragma unroll
      for (int j = 0; j < 32; j += 4) {
        float4 f = *(const float4*)(src + j);
        ta[j + 0] = (bf16_t)f.x;
        ta[j + 1] = (bf16_t)f.y;
        ta[j + 2] = (bf16_t)f.z;
        ta[j + 3] = (bf16_t)f.w;
      }
      uint4* dst = (uint4*)&As[tid][0];
      const uint4* s4 = (const uint4*)ta;
#pragma unroll
      for (int i = 0; i < 4; ++i) dst[i] = s4[i];
    }
    // ---- stage B tile transposed: X[kk+k][n0+t] -> Bs[t][k]
    //      (per-k reads are coalesced across the 128 threads)
    {
      const TB* src = Xb + (size_t)kk * NPOS + n0 + tid;
      if (kk + 32 < KDIM) __builtin_prefetch(src + (size_t)32 * NPOS, 0, 0);
      __align__(16) bf16_t tb[32];
#pragma unroll
      for (int k2 = 0; k2 < 32; ++k2)
        tb[k2] = (bf16_t)(float)src[(size_t)k2 * NPOS];
      uint4* dst = (uint4*)&Bs[tid][0];
      const uint4* s4 = (const uint4*)tb;
#pragma unroll
      for (int i = 0; i < 4; ++i) dst[i] = s4[i];
    }
    __syncthreads();

    // ---- B fragments (contiguous-K layout: lanes 0-15 = K 0..15,
    //      lanes 16-31 = K 16..31)
    FragU bfr[4];
#pragma unroll
    for (int ni = 0; ni < 4; ++ni) {
      const unsigned* rb = (const unsigned*)&Bs[wn + ni * 16 + l15][0];
#pragma unroll
      for (int j = 0; j < 8; ++j) bfr[ni].u[j] = rb[hi * 8 + j];
    }
    // ---- A fragments (contiguous-K layout: lanes 0-15 = K 0..15,
    //      lanes 16-31 = K 16..31) and WMMA accumulate
#pragma unroll
    for (int mi = 0; mi < 4; ++mi) {
      FragU a;
      const unsigned* rd = (const unsigned*)&As[wm + mi * 16 + l15][0];
#pragma unroll
      for (int j = 0; j < 4; ++j) a.u[j]     = rd[hi * 8 + j];
#pragma unroll
      for (int j = 0; j < 4; ++j) a.u[4 + j] = rd[hi * 8 + 4 + j];
#pragma unroll
      for (int ni = 0; ni < 4; ++ni) {
        acc[mi][ni] = __builtin_amdgcn_wmma_f32_16x16x32_bf16(
            false, a.v, false, bfr[ni].v, (short)0, acc[mi][ni], false, false);
      }
    }
    __syncthreads();
  }

  // ---- epilogue: C/D layout -> lanes 0-15 rows M..M+7, lanes 16-31 rows M+8..M+15
#pragma unroll
  for (int mi = 0; mi < 4; ++mi) {
#pragma unroll
    for (int ni = 0; ni < 4; ++ni) {
#pragma unroll
      for (int r = 0; r < 8; ++r) {
        const int row = m0 + wm + mi * 16 + hi * 8 + r;
        const int col = n0 + wn + ni * 16 + l15;
        float vv = acc[mi][ni][r];
        if constexpr (GELU)
          vv = 0.5f * vv * (1.0f + erff(vv * 0.70710678118654752f));
        Yb[(size_t)row * NPOS + col] = (TOUT)vv;
      }
    }
  }
}

// ---------------------------------------------------------------------------
// Attention core per (b,h): dots = (q^T k)*scale, softmax rows, out = v*att^T
// Block = 128 threads (4 waves) per slice. All matmuls via WMMA bf16.
// ---------------------------------------------------------------------------
__global__ __launch_bounds__(128)
void attn_kernel(const bf16_t* __restrict__ Q, const bf16_t* __restrict__ K,
                 const bf16_t* __restrict__ V, bf16_t* __restrict__ O)
{
  __shared__ __align__(16) bf16_t Qs[128][72];    // [i][d] (q transposed)
  __shared__ __align__(16) bf16_t Ks[128][72];    // [j][d] (k transposed)
  __shared__ __align__(16) bf16_t Vs[64][136];    // [d][j] (direct)
  __shared__ __align__(16) float  Ds[128][132];   // dots / exp scratch
  __shared__ __align__(16) bf16_t At[128][136];   // [i][j] att (bf16)

  const size_t base = (size_t)blockIdx.x * DD * NN;
  const int tid  = threadIdx.x;
  const int lane = tid & 31;
  const int wave = tid >> 5;
  const int hi   = lane >> 4;
  const int l15  = lane & 15;
  const v8f vzero = {0.f,0.f,0.f,0.f,0.f,0.f,0.f,0.f};

  // ---- V slice: contiguous bf16 copy -> async DMA to LDS (ASYNCcnt path).
  //      Per-lane LDS dest VGPR handles the padded row stride (136 elems).
  {
    const unsigned vsbase = (unsigned)(uintptr_t)&Vs[0][0];
#pragma unroll
    for (int p = 0; p < 8; ++p) {
      const int e = p * 1024 + tid * 8;           // 8 bf16 = 16B per lane
      const int d = e >> 7, n = e & 127;
      unsigned lds = vsbase + (unsigned)((d * 136 + n) * 2);
      unsigned long long ga = (unsigned long long)(uintptr_t)(V + base + e);
      asm volatile("global_load_async_to_lds_b128 %0, %1, off"
                   :: "v"(lds), "v"(ga) : "memory");
    }
  }

  // ---- Q,K transposed into LDS: b128 global loads + b16 scatter stores
  for (int p = 0; p < 8; ++p) {
    const int e = p * 1024 + tid * 8;
    const int d = e >> 7, n = e & 127;
    uint4 qv = *(const uint4*)(Q + base + e);
    uint4 kv = *(const uint4*)(K + base + e);
    const bf16_t* tq = (const bf16_t*)&qv;
    const bf16_t* tk = (const bf16_t*)&kv;
#pragma unroll
    for (int j = 0; j < 8; ++j) {
      Qs[n + j][d] = tq[j];
      Ks[n + j][d] = tk[j];
    }
  }
  asm volatile("s_wait_asynccnt 0x0" ::: "memory");
  __syncthreads();

  // ---- GEMM1: dots[i][j] = sum_d Qs[i][d]*Ks[j][d], wave = 64x64 tile
  {
    const int wm = (wave >> 1) * 64;   // i
    const int wn = (wave & 1) * 64;    // j
    v8f dacc[4][4];
#pragma unroll
    for (int i = 0; i < 4; ++i)
#pragma unroll
      for (int j = 0; j < 4; ++j) dacc[i][j] = vzero;

#pragma unroll
    for (int kk = 0; kk < DD; kk += 32) {
      const int kd = kk >> 1;          // dword offset into row
      FragU bfr[4];
#pragma unroll
      for (int ni = 0; ni < 4; ++ni) {
        const unsigned* rb = (const unsigned*)&Ks[wn + ni * 16 + l15][0];
#pragma unroll
        for (int j = 0; j < 8; ++j) bfr[ni].u[j] = rb[kd + hi * 8 + j];
      }
#pragma unroll
      for (int mi = 0; mi < 4; ++mi) {
        FragU a;
        const unsigned* rd = (const unsigned*)&Qs[wm + mi * 16 + l15][0];
#pragma unroll
        for (int j = 0; j < 4; ++j) a.u[j]     = rd[kd + hi * 8 + j];
#pragma unroll
        for (int j = 0; j < 4; ++j) a.u[4 + j] = rd[kd + hi * 8 + 4 + j];
#pragma unroll
        for (int ni = 0; ni < 4; ++ni)
          dacc[mi][ni] = __builtin_amdgcn_wmma_f32_16x16x32_bf16(
              false, a.v, false, bfr[ni].v, (short)0, dacc[mi][ni], false, false);
      }
    }
#pragma unroll
    for (int mi = 0; mi < 4; ++mi)
#pragma unroll
      for (int ni = 0; ni < 4; ++ni)
#pragma unroll
        for (int r = 0; r < 8; ++r)
          Ds[wm + mi * 16 + hi * 8 + r][wn + ni * 16 + l15] =
              dacc[mi][ni][r] * ATT_SCALE;
  }
  __syncthreads();

  // ---- softmax: one row per thread; att row written back as bf16 (b128)
  {
    float mx = -INFINITY;
#pragma unroll 4
    for (int j = 0; j < NN; ++j) mx = fmaxf(mx, Ds[tid][j]);
    float s = 0.f;
#pragma unroll 4
    for (int j = 0; j < NN; ++j) {
      float e = __expf(Ds[tid][j] - mx);
      s += e;
      Ds[tid][j] = e;
    }
    const float inv = 1.f / s;
    for (int j = 0; j < NN; j += 8) {
      __align__(16) bf16_t t8[8];
#pragma unroll
      for (int u2 = 0; u2 < 8; ++u2)
        t8[u2] = (bf16_t)(Ds[tid][j + u2] * inv);
      *(uint4*)&At[tid][j] = *(const uint4*)t8;
    }
  }
  __syncthreads();

  // ---- GEMM2: out[d][i] = sum_j Vs[d][j] * At[i][j]; wave owns 16 d-rows
  {
    const int d0 = wave * 16;
    v8f oacc[8];
#pragma unroll
    for (int i = 0; i < 8; ++i) oacc[i] = vzero;

#pragma unroll
    for (int kk = 0; kk < NN; kk += 32) {
      const int kd = kk >> 1;
      FragU a;
      const unsigned* rd = (const unsigned*)&Vs[d0 + l15][0];
#pragma unroll
      for (int j = 0; j < 4; ++j) a.u[j]     = rd[kd + hi * 8 + j];
#pragma unroll
      for (int j = 0; j < 4; ++j) a.u[4 + j] = rd[kd + hi * 8 + 4 + j];
#pragma unroll
      for (int ni = 0; ni < 8; ++ni) {
        FragU bfr;
        const unsigned* rb = (const unsigned*)&At[ni * 16 + l15][0];
#pragma unroll
        for (int j = 0; j < 8; ++j) bfr.u[j] = rb[kd + hi * 8 + j];
        oacc[ni] = __builtin_amdgcn_wmma_f32_16x16x32_bf16(
            false, a.v, false, bfr.v, (short)0, oacc[ni], false, false);
      }
    }
#pragma unroll
    for (int ni = 0; ni < 8; ++ni)
#pragma unroll
      for (int r = 0; r < 8; ++r) {
        const int d   = d0 + hi * 8 + r;
        const int col = ni * 16 + l15;
        O[base + (size_t)d * NN + col] = (bf16_t)oacc[ni][r];
      }
  }
}

// ---------------------------------------------------------------------------
extern "C" void kernel_launch(void* const* d_in, const int* in_sizes, int n_in,
                              void* d_out, int out_size, void* d_ws, size_t ws_size,
                              hipStream_t stream) {
  const float* lidar = (const float*)d_in[0];
  const float* cam   = (const float*)d_in[1];
  const float* Wq    = (const float*)d_in[2];
  const float* Wk    = (const float*)d_in[3];
  const float* Wv    = (const float*)d_in[4];
  const float* Wo    = (const float*)d_in[5];
  float* out = (float*)d_out;

  const size_t SLICE = (size_t)2 * 1024 * 64 * 128;  // elements per tensor
  bf16_t* qb = (bf16_t*)d_ws;
  bf16_t* kb = qb + SLICE;
  bf16_t* vb = kb + SLICE;
  bf16_t* ab = vb + SLICE;

  dim3 grid(NPOS / 128, MDIM / 128, 2);
  dim3 blk(128);

  proj_gemm<float, bf16_t, false><<<grid, blk, 0, stream>>>(Wq, lidar, qb);
  proj_gemm<float, bf16_t, false><<<grid, blk, 0, stream>>>(Wk, cam,   kb);
  proj_gemm<float, bf16_t, false><<<grid, blk, 0, stream>>>(Wv, cam,   vb);

  attn_kernel<<<dim3(2 * 1024), blk, 0, stream>>>(qb, kb, vb, ab);

  proj_gemm<bf16_t, float, true><<<grid, blk, 0, stream>>>(Wo, ab, out);
}
